// SparseAttnBottleneck_19688130085646
// MI455X (gfx1250) — matrix operand order repaired and checked by reading, not compiled, not run on hardware
//
#include <hip/hip_runtime.h>

#define VOC   65536
#define DIM   256
#define BATCH 4096
#define TOPK  32
#define NEG_MAX (-3.402823466e+38f)

typedef __attribute__((ext_vector_type(16))) __bf16 v16bf;
typedef __attribute__((ext_vector_type(8)))  float  v8f;

// ---------- helpers ----------

__device__ inline unsigned short f2bf(float f) {
  // round-to-nearest-even f32 -> bf16
  unsigned int x = __float_as_uint(f);
  unsigned int r = (x + 0x7FFFu + ((x >> 16) & 1u)) >> 16;
  return (unsigned short)r;
}

union AFrag { unsigned int u[8]; v16bf v; };

// A matrix 16x32 bf16 fragment: lanes 0-15 = rows M, lane half selects
// K=0..15 vs K=16..31; 16 consecutive bf16 along K (matches B fragment).
__device__ inline v16bf load_a_frag(const unsigned short* base, int row, int kk, int half) {
  return *(const v16bf*)(base + (size_t)row * DIM + kk + half * 16);
}

// B matrix 32x16 bf16 fragment: lane l16 = column N, lane half selects
// K=0..15 vs K=16..31; 16 consecutive bf16 along K (row-major source row).
__device__ inline v16bf load_b_frag(const unsigned short* base, int col, int kk, int half) {
  return *(const v16bf*)(base + (size_t)col * DIM + kk + half * 16);
}

// ---------- elementwise f32 -> bf16 conversion ----------

__global__ __launch_bounds__(256) void cvt_kernel(const float* __restrict__ src,
                                                  unsigned short* __restrict__ dst, int n) {
  int i = blockIdx.x * blockDim.x + threadIdx.x;
  if (i < n) dst[i] = f2bf(src[i]);
}

// ---------- projection GEMM: dst[m][n] = sum_k A[m][k]*W[n][k] + bias[n] ----------
// A: Mx256 bf16 row-major, W: 256x256 bf16 row-major (tiny, cache-resident).
// One wave computes a 64(M) x 32(N) tile; 8 waves/block stack along M.

template<int OUT_BF16>
__global__ __launch_bounds__(256) void proj_kernel(const unsigned short* __restrict__ A,
                                                   const unsigned short* __restrict__ W,
                                                   const float* __restrict__ bias,
                                                   void* __restrict__ dstv) {
  const int lane = threadIdx.x & 31;
  const int wave = threadIdx.x >> 5;
  const int half = lane >> 4;
  const int l16  = lane & 15;
  const int nbase = blockIdx.x * 32;
  const int mbase = (blockIdx.y * 8 + wave) * 64;

  v8f zero = {0.f, 0.f, 0.f, 0.f, 0.f, 0.f, 0.f, 0.f};
  v8f c[4][2];
#pragma unroll
  for (int mi = 0; mi < 4; ++mi)
#pragma unroll
    for (int ni = 0; ni < 2; ++ni) c[mi][ni] = zero;

  for (int kk = 0; kk < DIM; kk += 32) {
    v16bf a[4], b[2];
#pragma unroll
    for (int mi = 0; mi < 4; ++mi) a[mi] = load_a_frag(A, mbase + mi * 16 + l16, kk, half);
#pragma unroll
    for (int ni = 0; ni < 2; ++ni) b[ni] = load_b_frag(W, nbase + ni * 16 + l16, kk, half);
#pragma unroll
    for (int mi = 0; mi < 4; ++mi)
#pragma unroll
      for (int ni = 0; ni < 2; ++ni)
        c[mi][ni] = __builtin_amdgcn_wmma_f32_16x16x32_bf16(
            false, a[mi], false, b[ni], (short)0, c[mi][ni], false, false);
  }

#pragma unroll
  for (int mi = 0; mi < 4; ++mi)
#pragma unroll
    for (int ni = 0; ni < 2; ++ni) {
      int n = nbase + ni * 16 + l16;
      float bn = bias[n];
#pragma unroll
      for (int r = 0; r < 8; ++r) {
        int m = mbase + mi * 16 + r + half * 8;   // C/D layout: VGPR r -> M = r (+8 upper half)
        float val = c[mi][ni][r] + bn;
        if (OUT_BF16) ((unsigned short*)dstv)[(size_t)m * DIM + n] = f2bf(val);
        else          ((float*)dstv)[(size_t)m * DIM + n] = val;
      }
    }
}

// ---------- dots GEMM: dots[b][n] = sum_d q[b][d]*k[n][d] ----------
// qb: 4096x256 bf16, kb: 65536x256 bf16 (both L2-resident).
// All 8 waves of a block share nbase, so the 32-row B panel (32x256 bf16 =
// 16KB, contiguous in memory) is staged into LDS ONCE per block with the
// CDNA5 async global->LDS path (ASYNCcnt-tracked, no VGPR round trip),
// cutting redundant per-wave B reads 8x. A fragments stream from L2.
// One wave: 64(M) x 32(N) tile. grid = (VOC/32, BATCH/512).

__global__ __launch_bounds__(256) void dots_kernel(const unsigned short* __restrict__ qb,
                                                   const unsigned short* __restrict__ kb,
                                                   float* __restrict__ dots) {
  __shared__ __align__(32) unsigned short btile[32 * DIM];   // 16 KB B panel

  const int lane = threadIdx.x & 31;
  const int wave = threadIdx.x >> 5;
  const int half = lane >> 4;
  const int l16  = lane & 15;
  const int nbase = blockIdx.x * 32;
  const int mbase = (blockIdx.y * 8 + wave) * 64;

  // ---- async stage of the 16KB B panel: 1024 16-byte chunks, 4 per thread ----
  {
    const char* gsrc = (const char*)(kb + (size_t)nbase * DIM);  // rows nbase..nbase+31 contiguous
    int t = threadIdx.x;
#pragma unroll
    for (int i = 0; i < 4; ++i) {
      int chunk = t + i * 256;
      unsigned long long ga = (unsigned long long)(size_t)(gsrc + chunk * 16);
      unsigned lo = (unsigned)(size_t)btile + chunk * 16;      // low 32 bits = LDS offset
      asm volatile("global_load_async_to_lds_b128 %0, %1, off"
                   :: "v"(lo), "v"(ga) : "memory");
    }
    asm volatile("s_wait_asynccnt 0x0" ::: "memory");
    __syncthreads();
  }

  v8f zero = {0.f, 0.f, 0.f, 0.f, 0.f, 0.f, 0.f, 0.f};
  v8f c[4][2];
#pragma unroll
  for (int mi = 0; mi < 4; ++mi)
#pragma unroll
    for (int ni = 0; ni < 2; ++ni) c[mi][ni] = zero;

  for (int kk = 0; kk < DIM; kk += 32) {
    v16bf a[4], b[2];
#pragma unroll
    for (int mi = 0; mi < 4; ++mi) a[mi] = load_a_frag(qb, mbase + mi * 16 + l16, kk, half);
#pragma unroll
    for (int ni = 0; ni < 2; ++ni)
      b[ni] = *(const v16bf*)&btile[(size_t)(ni * 16 + l16) * DIM + kk + half * 16];
#pragma unroll
    for (int mi = 0; mi < 4; ++mi)
#pragma unroll
      for (int ni = 0; ni < 2; ++ni)
        c[mi][ni] = __builtin_amdgcn_wmma_f32_16x16x32_bf16(
            false, a[mi], false, b[ni], (short)0, c[mi][ni], false, false);
  }

#pragma unroll
  for (int mi = 0; mi < 4; ++mi)
#pragma unroll
    for (int ni = 0; ni < 2; ++ni) {
      int n = nbase + ni * 16 + l16;
#pragma unroll
      for (int r = 0; r < 8; ++r) {
        int m = mbase + mi * 16 + r + half * 8;
        dots[(size_t)m * VOC + n] = c[mi][ni][r];
      }
    }
}

// ---------- per-row top-32 + in-place masking ----------
// One block (128 threads) per row. Per-thread sorted top-32 lists in LDS,
// pairwise log-merge, then masked rewrite of the row (row re-read hits L2).

__global__ __launch_bounds__(128) void topk_mask_kernel(float* __restrict__ dots,
                                                        float* __restrict__ top_val,
                                                        int* __restrict__ top_ind) {
  __shared__ float sval[128 * TOPK];
  __shared__ int   sind[128 * TOPK];
  const int row = blockIdx.x;
  const int t = threadIdx.x;
  float* drow = dots + (size_t)row * VOC;

  float* L  = &sval[t * TOPK];
  int*   LI = &sind[t * TOPK];
#pragma unroll
  for (int i = 0; i < TOPK; ++i) { L[i] = NEG_MAX; LI[i] = -1; }

  for (int j = 0; j < VOC / 128; ++j) {
    int i = t + j * 128;
    float d = drow[i];
    if (d > L[TOPK - 1]) {
      int p = TOPK - 1;
      while (p > 0 && L[p - 1] < d) { L[p] = L[p - 1]; LI[p] = LI[p - 1]; --p; }
      L[p] = d; LI[p] = i;
    }
  }

  for (int step = 1; step < 128; step <<= 1) {
    __syncthreads();
    if ((t & (2 * step - 1)) == 0) {
      float ov[TOPK]; int oi[TOPK];
      float* Bv = &sval[(t + step) * TOPK];
      int*   Bi = &sind[(t + step) * TOPK];
      int ai = 0, bi = 0;
#pragma unroll
      for (int o = 0; o < TOPK; ++o) {
        float va = L[ai], vb = Bv[bi];
        if (va >= vb) { ov[o] = va; oi[o] = LI[ai]; ++ai; }
        else          { ov[o] = vb; oi[o] = Bi[bi]; ++bi; }
      }
#pragma unroll
      for (int o = 0; o < TOPK; ++o) { L[o] = ov[o]; LI[o] = oi[o]; }
    }
  }
  __syncthreads();

  if (t < TOPK) {
    top_val[(size_t)row * TOPK + t] = sval[t];
    top_ind[(size_t)row * TOPK + t] = sind[t];
  }
  float vk = sval[TOPK - 1];

  for (int j = 0; j < VOC / 128; ++j) {
    int i = t + j * 128;
    float d = drow[i];
    drow[i] = (d < vk) ? NEG_MAX : d;
  }
}

// ---------- softmax over top-32 + out = attn @ v ----------
// softmax(dots_masked) == softmax(top 32 values) since masked entries
// underflow to exp(...) = 0. One block (256 threads = DIM) per row.

__global__ __launch_bounds__(256) void softmax_out_kernel(const float* __restrict__ top_val,
                                                          const int* __restrict__ top_ind,
                                                          const float* __restrict__ v,
                                                          float* __restrict__ out) {
  __shared__ float w[TOPK];
  __shared__ int   id[TOPK];
  __shared__ float ssum;
  const int row = blockIdx.x;
  const int t = threadIdx.x;

  if (t < TOPK) {
    float m = top_val[(size_t)row * TOPK];          // max (sorted descending)
    w[t]  = __expf(top_val[(size_t)row * TOPK + t] - m);
    id[t] = top_ind[(size_t)row * TOPK + t];
  }
  __syncthreads();
  if (t == 0) {
    float s = 0.f;
#pragma unroll
    for (int i = 0; i < TOPK; ++i) s += w[i];
    ssum = s;
  }
  __syncthreads();
  float inv = 1.0f / ssum;

  float acc = 0.f;
#pragma unroll
  for (int i = 0; i < TOPK; ++i) acc += w[i] * v[(size_t)id[i] * DIM + t];
  out[(size_t)row * DIM + t] = acc * inv;
}

// ---------- host launcher ----------

extern "C" void kernel_launch(void* const* d_in, const int* in_sizes, int n_in,
                              void* d_out, int out_size, void* d_ws, size_t ws_size,
                              hipStream_t stream) {
  (void)in_sizes; (void)n_in; (void)out_size; (void)ws_size;

  const float* x        = (const float*)d_in[0];
  const float* codebook = (const float*)d_in[1];
  const float* Wq       = (const float*)d_in[2];
  const float* bq       = (const float*)d_in[3];
  const float* Wk       = (const float*)d_in[4];
  const float* bk       = (const float*)d_in[5];
  const float* Wv       = (const float*)d_in[6];
  const float* bv       = (const float*)d_in[7];

  // output layout: out | top_value | top_ind | dots_masked
  float* out      = (float*)d_out;
  float* top_val  = out + (size_t)BATCH * DIM;
  int*   top_ind  = (int*)(top_val + (size_t)BATCH * TOPK);
  float* dots     = top_val + 2 * (size_t)BATCH * TOPK;

  // workspace layout (bf16 operand copies + f32 v)
  char* w = (char*)d_ws;
  unsigned short* xb  = (unsigned short*)w; w += (size_t)BATCH * DIM * 2;
  unsigned short* cbb = (unsigned short*)w; w += (size_t)VOC * DIM * 2;
  unsigned short* wqb = (unsigned short*)w; w += (size_t)DIM * DIM * 2;
  unsigned short* wkb = (unsigned short*)w; w += (size_t)DIM * DIM * 2;
  unsigned short* wvb = (unsigned short*)w; w += (size_t)DIM * DIM * 2;
  unsigned short* qb  = (unsigned short*)w; w += (size_t)BATCH * DIM * 2;
  unsigned short* kb  = (unsigned short*)w; w += (size_t)VOC * DIM * 2;
  float*          vf  = (float*)w;          w += (size_t)VOC * DIM * 4;

  // 1) convert operands to bf16
  cvt_kernel<<<(BATCH * DIM) / 256, 256, 0, stream>>>(x, xb, BATCH * DIM);
  cvt_kernel<<<(VOC * DIM) / 256, 256, 0, stream>>>(codebook, cbb, VOC * DIM);
  cvt_kernel<<<(DIM * DIM) / 256, 256, 0, stream>>>(Wq, wqb, DIM * DIM);
  cvt_kernel<<<(DIM * DIM) / 256, 256, 0, stream>>>(Wk, wkb, DIM * DIM);
  cvt_kernel<<<(DIM * DIM) / 256, 256, 0, stream>>>(Wv, wvb, DIM * DIM);

  // 2) projections (q,k kept bf16 for the dots GEMM; v kept f32 for output)
  proj_kernel<1><<<dim3(DIM / 32, BATCH / 512), 256, 0, stream>>>(xb, wqb, bq, qb);
  proj_kernel<1><<<dim3(DIM / 32, VOC / 512),   256, 0, stream>>>(cbb, wkb, bk, kb);
  proj_kernel<0><<<dim3(DIM / 32, VOC / 512),   256, 0, stream>>>(cbb, wvb, bv, vf);

  // 3) dots = q @ k^T  (writes raw dots into the dots_masked output region)
  dots_kernel<<<dim3(VOC / 32, BATCH / 512), 256, 0, stream>>>(qb, kb, dots);

  // 4) per-row top-32 + in-place masking
  topk_mask_kernel<<<BATCH, 128, 0, stream>>>(dots, top_val, top_ind);

  // 5) softmax over survivors + out = attn @ v
  softmax_out_kernel<<<BATCH, 256, 0, stream>>>(top_val, top_ind, vf, out);
}
